// Model_24292335026637
// MI455X (gfx1250) — compile-verified, matrix operand order checked by reading, not run on hardware
//
#include <hip/hip_runtime.h>
#include <hip/hip_bf16.h>

#define BB 16
#define LL 4096
#define DD 256
#define NN 64
#define NLAY 6
#define BLROWS (BB*LL)   // 65536 token rows

typedef __bf16 v16bf __attribute__((ext_vector_type(16)));
typedef float  v8f   __attribute__((ext_vector_type(8)));
typedef unsigned u32x4 __attribute__((ext_vector_type(4)));
typedef unsigned u32x8 __attribute__((ext_vector_type(8)));

// ---------- helpers ----------
__device__ __forceinline__ unsigned f2bf(float f) {           // RNE (weights, one-time)
    unsigned u = __float_as_uint(f);
    unsigned r = ((u >> 16) & 1u) + 0x7FFFu;
    return (u + r) >> 16;
}

// pack two f32 -> bf16x2 in ONE v_perm_b32 (truncation; activations)
__device__ __forceinline__ unsigned pack_bf16x2(float lo, float hi) {
    return __builtin_amdgcn_perm(__float_as_uint(hi), __float_as_uint(lo), 0x07060302u);
}

// LDS byte offset of a __shared__ pointer (addrspace(3) cast)
__device__ __forceinline__ unsigned ldsoff(void* p) {
    return (unsigned)(unsigned long long)(__attribute__((address_space(3))) void*)p;
}

__device__ __forceinline__ float gelu_f(float x) {
    const float c = 0.7978845608028654f;
    return 0.5f * x * (1.0f + tanhf(c * (x + 0.044715f * x * x * x)));
}

__device__ __forceinline__ float block_sum256(float v, float* red, int c) {
    red[c] = v;
    __syncthreads();
#pragma unroll
    for (int s = 128; s > 0; s >>= 1) {
        if (c < s) red[c] += red[c + s];
        __syncthreads();
    }
    float r = red[0];
    __syncthreads();
    return r;
}

// Tensor Data Mover: 2D tile load Global->LDS (D# per CDNA5 ISA; 2-group form)
__device__ __forceinline__ void tdm_load_2d(unsigned lds_addr, const void* gaddr,
                                            unsigned tile_d0, unsigned tile_d1,
                                            unsigned stride_d0, unsigned pad_en,
                                            unsigned pad_interval_code,
                                            unsigned pad_amount_code) {
    unsigned long long ga = (unsigned long long)(uintptr_t)gaddr;
    const unsigned td0 = 1u << 20, td1 = 1u << 20;  // huge tensor dims: no OOB clip
    u32x4 g0;
    g0.x = 1u;                                            // count=1, user descriptor
    g0.y = lds_addr;                                      // LDS byte address
    g0.z = (unsigned)ga;                                  // global_addr[31:0]
    g0.w = (unsigned)((ga >> 32) & 0x1FFFFFFu) | (2u << 30); // addr[56:32], type=2
    u32x8 g1;
    g1[0] = (2u << 16) | (pad_en << 20) | (pad_interval_code << 22)
          | (pad_amount_code << 25);                      // data_size=4B, padding cfg
    g1[1] = (td0 & 0xFFFFu) << 16;                        // tensor_dim0[15:0]
    g1[2] = (td0 >> 16) | ((td1 & 0xFFFFu) << 16);        // tdim0[31:16], tdim1[15:0]
    g1[3] = (td1 >> 16) | (tile_d0 << 16);                // tdim1[31:16], tile_dim0
    g1[4] = tile_d1;                                      // tile_dim1, tile_dim2=0
    g1[5] = stride_d0;                                    // tensor_dim0_stride[31:0]
    g1[6] = 0u;
    g1[7] = 0u;
    asm volatile("tensor_load_to_lds %0, %1" :: "s"(g0), "s"(g1) : "memory");
}

// ---------- kernel 1: embedding gather + layernorm ----------
__global__ void embed_ln_k(const int* __restrict__ x, const float* __restrict__ emb,
                           const float* __restrict__ g, const float* __restrict__ b,
                           float* __restrict__ out) {
    __shared__ float red[DD];
    size_t row = blockIdx.x;
    int c = threadIdx.x;
    int tok = x[row];
    float v = emb[(size_t)tok * DD + c];
    float mean = block_sum256(v, red, c) * (1.0f / DD);
    float dv = v - mean;
    float var = block_sum256(dv * dv, red, c) * (1.0f / DD);
    out[row * DD + c] = dv * rsqrtf(var + 1e-5f) * g[c] + b[c];
}

// ---------- kernel 2: per-layer layernorm ----------
__global__ void ln_rows_k(const float* __restrict__ in, const float* __restrict__ g,
                          const float* __restrict__ b, float* __restrict__ out) {
    __shared__ float red[DD];
    size_t row = blockIdx.x;
    int c = threadIdx.x;
    float v = in[row * DD + c];
    float mean = block_sum256(v, red, c) * (1.0f / DD);
    float dv = v - mean;
    float var = block_sum256(dv * dv, red, c) * (1.0f / DD);
    out[row * DD + c] = dv * rsqrtf(var + 1e-5f) * g[c] + b[c];
}

// ---------- kernel 3: ZOH discretization coefficients ----------
__global__ void s4d_coeff_k(const float* __restrict__ logdt, const float* __restrict__ Alog,
                            const float* __restrict__ Bp, const float* __restrict__ Cp,
                            float* __restrict__ Abar, float* __restrict__ CB) {
    int idx = blockIdx.x * 256 + threadIdx.x;  // < D*N
    int d = idx >> 6;                          // / N
    float dt = expf(logdt[d]);
    float A  = -expf(Alog[idx]);
    float ab = expf(dt * A);
    float bb = (ab - 1.0f) / A * Bp[idx];
    Abar[idx] = ab;
    CB[idx]   = Cp[idx] * bb;
}

// ---------- kernel 4: S4D diagonal scan + gelu + residual ----------
__global__ void s4d_scan_k(const float* __restrict__ hn, float* __restrict__ h,
                           const float2* __restrict__ Abar, const float2* __restrict__ CB,
                           const float* __restrict__ Dp) {
    __shared__ float tile[8][32][33];
    int lane = threadIdx.x & 31;
    int w    = threadIdx.x >> 5;
    int gw   = blockIdx.x * 8 + w;
    int bb   = gw >> 8;
    int d    = gw & (DD - 1);
    float2 a2 = Abar[d * (NN / 2) + lane];
    float2 c2 = CB[d * (NN / 2) + lane];
    float dskip = Dp[d];
    float s0 = 0.f, s1 = 0.f;
    size_t base0 = (size_t)bb * LL * DD + d;

    for (int c0 = 0; c0 < LL; c0 += 32) {
        size_t idx = base0 + (size_t)(c0 + lane) * DD;
        float u  = hn[idx];
        float hv = h[idx];
        if (c0 + 32 < LL)
            __builtin_prefetch(&hn[base0 + (size_t)(c0 + 32 + lane) * DD], 0, 1);
        for (int j = 0; j < 32; j++) {
            float uj = __shfl(u, j, 32);
            s0 = fmaf(a2.x, s0, uj);
            s1 = fmaf(a2.y, s1, uj);
            tile[w][lane][j] = fmaf(c2.x, s0, c2.y * s1);
        }
        __syncthreads();
        float p = 0.f;
#pragma unroll
        for (int i = 0; i < 32; i++) p += tile[w][i][lane];
        __syncthreads();
        h[idx] = hv + gelu_f(p + u * dskip);
    }
}

// ---------- kernel 5: pack W into WMMA B-fragment bf16 layout (RNE) ----------
__global__ void pack_w_k(const float* __restrict__ W, unsigned* __restrict__ Wf) {
    int tid  = blockIdx.x * 256 + threadIdx.x;  // < 4096
    int lane = tid & 31;
    int rest = tid >> 5;
    int ks   = rest & 7;
    int nt   = rest >> 3;
    int hi   = lane >> 4;
    int ncol = nt * 16 + (lane & 15);
#pragma unroll
    for (int j = 0; j < 8; j++) {
        int k = ks * 32 + ((j & 4) ? 16 : 0) + hi * 8 + (j & 3) * 2;
        unsigned lo   = f2bf(W[(size_t)k * DD + ncol]);
        unsigned hi16 = f2bf(W[(size_t)(k + 1) * DD + ncol]);
        Wf[(size_t)tid * 8 + j] = lo | (hi16 << 16);
    }
}

// ---------- kernel 6: FFN GEMM, async-LDS staged A + pipelined B + WMMA ----------
#define AST 36   // padded row stride (floats): 16B-aligned, conflict-free frag reads
__global__ void ffn_gemm_k(const float* __restrict__ Amat, const unsigned* __restrict__ Wf,
                           const float* __restrict__ bias, const float* __restrict__ resid,
                           float* __restrict__ Out, int addres) {
    __shared__ float abuf[2][64 * AST];
    int tid  = threadIdx.x;
    int wid  = tid >> 5;
    int lane = tid & 31;
    int wm = wid >> 1, wn = wid & 1;
    int bm = blockIdx.x >> 1, bn = blockIdx.x & 1;
    int row0b = bm * 64;
    int row0  = row0b + wm * 16;
    int col0  = bn * 128 + wn * 64;
    int mrow = lane & 15, hi = lane >> 4;
    int nt0 = col0 >> 4;

    auto stage = [&](int s, int ks) {
#pragma unroll
        for (int q = 0; q < 2; q++) {
            int qi  = q * 256 + tid;            // 512 quads of 4 floats
            int row = qi >> 3, qc = qi & 7;
            const float* gp = Amat + (size_t)(row0b + row) * DD + ks * 32 + qc * 4;
            unsigned loff = ldsoff(&abuf[s][row * AST + qc * 4]);
            unsigned long long ga = (unsigned long long)(uintptr_t)gp;
            asm volatile("global_load_async_to_lds_b128 %0, %1, off"
                         :: "v"(loff), "v"(ga) : "memory");
        }
    };

    v8f zero = {0.f, 0.f, 0.f, 0.f, 0.f, 0.f, 0.f, 0.f};
    v8f acc[4] = {zero, zero, zero, zero};

    stage(0, 0);

    for (int ks = 0; ks < 8; ks++) {
        asm volatile("s_wait_asynccnt 0x0" ::: "memory");  // own stage(ks) done
        __syncthreads();                                    // all waves' stages visible
        if (ks < 7) stage((ks + 1) & 1, ks + 1);            // overlap next slab

        union { unsigned u[8]; v16bf v; } af;
        const float* ab = &abuf[ks & 1][(wm * 16 + mrow) * AST];
#pragma unroll
        for (int j = 0; j < 8; j++) {
            int koff = ((j & 4) ? 16 : 0) + hi * 8 + (j & 3) * 2;
            float2 p = *(const float2*)(ab + koff);
            af.u[j] = pack_bf16x2(p.x, p.y);
        }

        const uint4* q0p = (const uint4*)(Wf + ((size_t)(nt0 * 8 + ks) * 32 + lane) * 8);
        uint4 c0 = q0p[0], c1 = q0p[1];
#pragma unroll
        for (int t = 0; t < 4; t++) {
            uint4 n0, n1;
            if (t < 3) {
                const uint4* qn = (const uint4*)(Wf + ((size_t)((nt0 + t + 1) * 8 + ks) * 32 + lane) * 8);
                n0 = qn[0]; n1 = qn[1];
            }
            union { unsigned u[8]; v16bf v; } bf;
            bf.u[0] = c0.x; bf.u[1] = c0.y; bf.u[2] = c0.z; bf.u[3] = c0.w;
            bf.u[4] = c1.x; bf.u[5] = c1.y; bf.u[6] = c1.z; bf.u[7] = c1.w;
            acc[t] = __builtin_amdgcn_wmma_f32_16x16x32_bf16(
                false, af.v, false, bf.v, (short)0, acc[t], false, false);
            if (t < 3) { c0 = n0; c1 = n1; }
        }
    }

#pragma unroll
    for (int t = 0; t < 4; t++) {
        int col = col0 + t * 16 + (lane & 15);
        float bcol = bias[col];
#pragma unroll
        for (int r = 0; r < 8; r++) {
            int row = row0 + hi * 8 + r;
            size_t o = (size_t)row * DD + col;
            float v = acc[t][r] + bcol;
            if (addres) v += resid[o];
            Out[o] = gelu_f(v);
        }
    }
}

// ---------- kernel 7: mean pool via TDM-staged slabs, then [D]x[D,2] fc ----------
__global__ void pool_fc_k(const float* __restrict__ h, const float* __restrict__ fcW,
                          const float* __restrict__ fcb, float* __restrict__ out) {
    __shared__ float buf[2][32 * DD];     // 2 x 32KB slabs staged by TDM
    __shared__ float r0[DD], r1[DD];
    int b = blockIdx.x;
    int d = threadIdx.x;

    auto stage = [&](int s, int c) {
        if (threadIdx.x < 32) {           // single wave issues the descriptor
            tdm_load_2d(ldsoff(&buf[s][0]),
                        h + ((size_t)b * LL + (size_t)c * 32) * DD,
                        /*tile_d0=*/32 * DD, /*tile_d1=*/1,
                        /*stride_d0=*/32 * DD, /*pad_en=*/0, 0, 0);
        }
    };

    float acc = 0.f;
    stage(0, 0);
    for (int c = 0; c < LL / 32; c++) {
        __builtin_amdgcn_s_wait_tensorcnt(0);   // issuing wave's TENSORcnt drains
        __syncthreads();                        // slab visible to whole block
        if (c < LL / 32 - 1) stage((c + 1) & 1, c + 1);  // overlap next slab
#pragma unroll
        for (int l = 0; l < 32; l++) acc += buf[c & 1][l * DD + d];
    }
    float pooled = acc * (1.0f / LL);
    r0[d] = pooled * fcW[d * 2 + 0];
    r1[d] = pooled * fcW[d * 2 + 1];
    __syncthreads();
    for (int st = DD / 2; st > 0; st >>= 1) {
        if (d < st) { r0[d] += r0[d + st]; r1[d] += r1[d + st]; }
        __syncthreads();
    }
    if (d == 0) {
        out[b * 2 + 0] = r0[0] + fcb[0];
        out[b * 2 + 1] = r1[0] + fcb[1];
    }
}

// ---------- host launcher ----------
extern "C" void kernel_launch(void* const* d_in, const int* in_sizes, int n_in,
                              void* d_out, int out_size, void* d_ws, size_t ws_size,
                              hipStream_t stream) {
    (void)in_sizes; (void)n_in; (void)out_size; (void)ws_size;
    const int*   x     = (const int*)  d_in[0];
    const float* emb   = (const float*)d_in[1];
    const float* emb_g = (const float*)d_in[2];
    const float* emb_b = (const float*)d_in[3];
    const float* logdt = (const float*)d_in[4];   // [6,256]
    const float* Alog  = (const float*)d_in[5];   // [6,256,64]
    const float* Bp    = (const float*)d_in[6];
    const float* Cp    = (const float*)d_in[7];
    const float* Dp    = (const float*)d_in[8];   // [6,256]
    const float* ln_g  = (const float*)d_in[9];
    const float* ln_b  = (const float*)d_in[10];
    const float* W1    = (const float*)d_in[11];
    const float* b1    = (const float*)d_in[12];
    const float* W2    = (const float*)d_in[13];
    const float* b2    = (const float*)d_in[14];
    const float* fcW   = (const float*)d_in[15];
    const float* fcb   = (const float*)d_in[16];

    char* ws = (char*)d_ws;
    const size_t actBytes = (size_t)BLROWS * DD * 4;          // 64 MB
    float*    h    = (float*)(ws);
    float*    t1   = (float*)(ws + actBytes);                 // hn per layer, then FFN mid
    float*    Abar = (float*)(ws + 2 * actBytes);
    float*    CBp  = Abar + DD * NN;
    unsigned* Wf1  = (unsigned*)(CBp + DD * NN);
    unsigned* Wf2  = Wf1 + 16 * 8 * 32 * 8;

    pack_w_k<<<16, 256, 0, stream>>>(W1, Wf1);
    pack_w_k<<<16, 256, 0, stream>>>(W2, Wf2);

    embed_ln_k<<<BLROWS, 256, 0, stream>>>(x, emb, emb_g, emb_b, h);

    for (int i = 0; i < NLAY; i++) {
        s4d_coeff_k<<<(DD * NN) / 256, 256, 0, stream>>>(
            logdt + (size_t)i * DD, Alog + (size_t)i * DD * NN,
            Bp + (size_t)i * DD * NN, Cp + (size_t)i * DD * NN, Abar, CBp);
        ln_rows_k<<<BLROWS, 256, 0, stream>>>(h, ln_g + (size_t)i * DD,
                                              ln_b + (size_t)i * DD, t1);
        s4d_scan_k<<<(BB * DD) / 8, 256, 0, stream>>>(
            t1, h, (const float2*)Abar, (const float2*)CBp, Dp + (size_t)i * DD);
    }

    ffn_gemm_k<<<(BLROWS / 64) * 2, 256, 0, stream>>>(h, Wf1, b1, nullptr, t1, 0);
    ffn_gemm_k<<<(BLROWS / 64) * 2, 256, 0, stream>>>(t1, Wf2, b2, h, h, 1);

    pool_fc_k<<<BB, 256, 0, stream>>>(h, fcW, fcb, (float*)d_out);
}
